// MultiHeadRankawareAttentionV2_80642305949921
// MI455X (gfx1250) — matrix operand order repaired and checked by reading, not compiled, not run on hardware
//
#include <hip/hip_runtime.h>
#include <cstdint>
#include <cstddef>

typedef __bf16 bf16;
typedef __attribute__((ext_vector_type(16))) __bf16 v16bf;
typedef __attribute__((ext_vector_type(8)))  float  v8f;

#define LN_EPS 1e-5f

// ---------------------------------------------------------------------------
// WMMA fragment helpers (16x16x32 bf16, wave32). Layouts used here (contiguous k order):
//   A (16x32, row-major src): lane L<16 -> row L, K={0..15}; L>=16 -> K={16..31}
//   B (32x16, B^T source i.e. W[N,K] row-major): lane = col n (&15), K-half = lane>>4
//   C/D: col = lane&15, rows = vgpr + 8*(lane>>4)
// frag_*_at take the exact per-lane base address (precomputed once); the k-step
// is applied as a constant element offset so it folds into the load immediate.
// ---------------------------------------------------------------------------
__device__ __forceinline__ v16bf frag_a_at(const bf16* __restrict__ p) {
  v16bf r;
#pragma unroll
  for (int t = 0; t < 8; ++t) { r[t] = p[t]; r[t + 8] = p[t + 8]; }
  return r;
}

__device__ __forceinline__ v16bf frag_b_at(const bf16* __restrict__ p) {
  v16bf r;
#pragma unroll
  for (int t = 0; t < 16; ++t) r[t] = p[t];
  return r;
}

__device__ __forceinline__ const bf16* lane_base_a(const bf16* A, int lda, int lane) {
  return A + (size_t)(lane & 15) * lda + (lane >> 4) * 16;
}
__device__ __forceinline__ const bf16* lane_base_b(const bf16* B, int ldb, int lane) {
  return B + (size_t)(lane & 15) * ldb + (lane >> 4) * 16;
}

__device__ __forceinline__ v16bf load_frag_a(const bf16* __restrict__ A, int lda, int lane) {
  return frag_a_at(lane_base_a(A, lda, lane));
}
__device__ __forceinline__ v16bf load_frag_b(const bf16* __restrict__ B, int ldb, int lane) {
  return frag_b_at(lane_base_b(B, ldb, lane));
}

__device__ __forceinline__ v8f wmma_bf16(v16bf a, v16bf b, v8f c) {
  return __builtin_amdgcn_wmma_f32_16x16x32_bf16(false, a, false, b, (short)0, c, false, false);
}

// ---------------------------------------------------------------------------
// fp32 -> bf16 convert with zero-padded tail (grid-stride)
// ---------------------------------------------------------------------------
__global__ void cvt_f32_to_bf16(const float* __restrict__ src, bf16* __restrict__ dst,
                                int n_src, int n_total) {
  int i = blockIdx.x * blockDim.x + threadIdx.x;
  const int stride = gridDim.x * blockDim.x;
  for (; i < n_total; i += stride) dst[i] = (i < n_src) ? (bf16)src[i] : (bf16)0.0f;
}

// ---------------------------------------------------------------------------
// C[M,N] = A[M,K] @ B[N,K]^T + bias  (bf16 in, fp32 accumulate/out)
// block = 256 threads = 8 waves; each wave owns a 32x64 output strip
// (2 M-tiles x 4 N-tiles -> 8 accumulators, B fragments reused twice).
// Ping-pong software pipeline (unroll-2): even/odd fragment sets are reloaded
// in place right after being consumed, so loads for K-step k+2 overlap the
// 8 WMMAs of step k+1 with zero register-rotation moves. Per-lane fragment
// addresses are computed once and bumped by a constant, so the k-step offset
// folds into the load immediates. Requires M%32==0, N%64==0, K%64==0.
// ---------------------------------------------------------------------------
__global__ void __launch_bounds__(256, 1)
gemm_bt_wmma(const bf16* __restrict__ A, const bf16* __restrict__ B,
             const float* __restrict__ bias, float* __restrict__ C,
             int M, int N, int K) {
  const int lane = threadIdx.x & 31;
  const int wave = threadIdx.x >> 5;
  const int tiles_n = N >> 6;
  const int strips_m = M >> 5;
  const int g = blockIdx.x * (blockDim.x >> 5) + wave;
  const int ms = g / tiles_n;
  const int ng = g - ms * tiles_n;
  if (ms >= strips_m) return;

  // exact per-lane fragment base addresses (computed once)
  const bf16* pa0 = lane_base_a(A + (size_t)ms * 32 * K, K, lane);
  const bf16* pa1 = pa0 + (size_t)16 * K;
  const bf16* pb0 = lane_base_b(B + (size_t)ng * 64 * K, K, lane);
  const bf16* pb1 = pb0 + (size_t)16 * K;
  const bf16* pb2 = pb0 + (size_t)32 * K;
  const bf16* pb3 = pb0 + (size_t)48 * K;

  v8f a00 = {}, a01 = {}, a02 = {}, a03 = {};
  v8f a10 = {}, a11 = {}, a12 = {}, a13 = {};

  // prologue: even set = K-step 0, odd set = K-step 32
  v16bf ea0 = frag_a_at(pa0),      ea1 = frag_a_at(pa1);
  v16bf eb0 = frag_b_at(pb0),      eb1 = frag_b_at(pb1);
  v16bf eb2 = frag_b_at(pb2),      eb3 = frag_b_at(pb3);
  v16bf oa0 = frag_a_at(pa0 + 32), oa1 = frag_a_at(pa1 + 32);
  v16bf ob0 = frag_b_at(pb0 + 32), ob1 = frag_b_at(pb1 + 32);
  v16bf ob2 = frag_b_at(pb2 + 32), ob3 = frag_b_at(pb3 + 32);

  const int iters = (K >> 6) - 1;  // K/64 double-steps, minus the epilogue
  for (int it = 0; it < iters; ++it) {
    if (it + 2 < iters) {  // bounded speculative prefetch (global_prefetch_b8)
      __builtin_prefetch(pa0 + 192, 0, 1);
      __builtin_prefetch(pb0 + 192, 0, 1);
    }
    // consume even set, then refill it from the next double-step
    a00 = wmma_bf16(ea0, eb0, a00);
    a01 = wmma_bf16(ea0, eb1, a01);
    a02 = wmma_bf16(ea0, eb2, a02);
    a03 = wmma_bf16(ea0, eb3, a03);
    a10 = wmma_bf16(ea1, eb0, a10);
    a11 = wmma_bf16(ea1, eb1, a11);
    a12 = wmma_bf16(ea1, eb2, a12);
    a13 = wmma_bf16(ea1, eb3, a13);
    pa0 += 64; pa1 += 64; pb0 += 64; pb1 += 64; pb2 += 64; pb3 += 64;
    ea0 = frag_a_at(pa0); ea1 = frag_a_at(pa1);
    eb0 = frag_b_at(pb0); eb1 = frag_b_at(pb1);
    eb2 = frag_b_at(pb2); eb3 = frag_b_at(pb3);
    // consume odd set (loads above are in flight), refill it
    a00 = wmma_bf16(oa0, ob0, a00);
    a01 = wmma_bf16(oa0, ob1, a01);
    a02 = wmma_bf16(oa0, ob2, a02);
    a03 = wmma_bf16(oa0, ob3, a03);
    a10 = wmma_bf16(oa1, ob0, a10);
    a11 = wmma_bf16(oa1, ob1, a11);
    a12 = wmma_bf16(oa1, ob2, a12);
    a13 = wmma_bf16(oa1, ob3, a13);
    oa0 = frag_a_at(pa0 + 32); oa1 = frag_a_at(pa1 + 32);
    ob0 = frag_b_at(pb0 + 32); ob1 = frag_b_at(pb1 + 32);
    ob2 = frag_b_at(pb2 + 32); ob3 = frag_b_at(pb3 + 32);
  }
  // epilogue: last even + odd sets
  a00 = wmma_bf16(ea0, eb0, a00);
  a01 = wmma_bf16(ea0, eb1, a01);
  a02 = wmma_bf16(ea0, eb2, a02);
  a03 = wmma_bf16(ea0, eb3, a03);
  a10 = wmma_bf16(ea1, eb0, a10);
  a11 = wmma_bf16(ea1, eb1, a11);
  a12 = wmma_bf16(ea1, eb2, a12);
  a13 = wmma_bf16(ea1, eb3, a13);
  a00 = wmma_bf16(oa0, ob0, a00);
  a01 = wmma_bf16(oa0, ob1, a01);
  a02 = wmma_bf16(oa0, ob2, a02);
  a03 = wmma_bf16(oa0, ob3, a03);
  a10 = wmma_bf16(oa1, ob0, a10);
  a11 = wmma_bf16(oa1, ob1, a11);
  a12 = wmma_bf16(oa1, ob2, a12);
  a13 = wmma_bf16(oa1, ob3, a13);

  const int r0 = 8 * (lane >> 4);
  const int nc = ng * 64 + (lane & 15);
  const float bv0 = bias ? bias[nc]      : 0.0f;
  const float bv1 = bias ? bias[nc + 16] : 0.0f;
  const float bv2 = bias ? bias[nc + 32] : 0.0f;
  const float bv3 = bias ? bias[nc + 48] : 0.0f;
  {
    const int row = ms * 32 + r0;
#pragma unroll
    for (int r = 0; r < 8; ++r) {
      float* cp = C + (size_t)(row + r) * N + nc;
      cp[0]  = a00[r] + bv0;
      cp[16] = a01[r] + bv1;
      cp[32] = a02[r] + bv2;
      cp[48] = a03[r] + bv3;
    }
  }
  {
    const int row = ms * 32 + 16 + r0;
#pragma unroll
    for (int r = 0; r < 8; ++r) {
      float* cp = C + (size_t)(row + r) * N + nc;
      cp[0]  = a10[r] + bv0;
      cp[16] = a11[r] + bv1;
      cp[32] = a12[r] + bv2;
      cp[48] = a13[r] + bv3;
    }
  }
}

// ---------------------------------------------------------------------------
// q path: Linear(768->768) + bias + LayerNorm, write bf16 padded to 16 rows
// (rows 8..15 zero so WMMA M-padding is clean). grid=16 blocks, 256 threads.
// ---------------------------------------------------------------------------
__global__ void qproj_ln(const float* __restrict__ q_x, const float* __restrict__ q_w,
                         const float* __restrict__ q_b, const float* __restrict__ lnw,
                         const float* __restrict__ lnb, bf16* __restrict__ qpad) {
  const int row = blockIdx.x;
  const int tid = threadIdx.x;
  if (row >= 8) {
    for (int c = tid; c < 768; c += 256) qpad[(size_t)row * 768 + c] = (bf16)0.0f;
    return;
  }
  __shared__ float rowbuf[768];
  __shared__ float red[256];
  for (int c = tid; c < 768; c += 256) {
    float s = q_b[c];
    const float* xr = q_x + (size_t)row * 768;
    const float* wr = q_w + (size_t)c * 768;
    for (int k = 0; k < 768; ++k) s += xr[k] * wr[k];
    rowbuf[c] = s;
  }
  __syncthreads();
  float part = 0.0f;
  for (int c = tid; c < 768; c += 256) part += rowbuf[c];
  red[tid] = part; __syncthreads();
  for (int s = 128; s > 0; s >>= 1) { if (tid < s) red[tid] += red[tid + s]; __syncthreads(); }
  const float mean = red[0] / 768.0f;
  __syncthreads();
  part = 0.0f;
  for (int c = tid; c < 768; c += 256) { const float d = rowbuf[c] - mean; part += d * d; }
  red[tid] = part; __syncthreads();
  for (int s = 128; s > 0; s >>= 1) { if (tid < s) red[tid] += red[tid + s]; __syncthreads(); }
  const float rstd = rsqrtf(red[0] / 768.0f + LN_EPS);
  for (int c = tid; c < 768; c += 256)
    qpad[(size_t)row * 768 + c] = (bf16)((rowbuf[c] - mean) * rstd * lnw[c] + lnb[c]);
}

// ---------------------------------------------------------------------------
// kv path post-GEMM: LayerNorm + positional bias, producing
//   kvA[i,p,:] = LN(kv[i,p]) + pos[p]      (first-half tokens)
//   kvB[i,p,:] = LN(kv[i,p]) + pos[197+p]  (second-half tokens)
// padded to 208 tokens per batch (zeros). grid = 16*208 blocks.
// ---------------------------------------------------------------------------
__global__ void ln_pos(const float* __restrict__ tmp, const float* __restrict__ lnw,
                       const float* __restrict__ lnb, const float* __restrict__ pos,
                       bf16* __restrict__ kvA, bf16* __restrict__ kvB) {
  const int row = blockIdx.x;            // i*208 + p
  const int i = row / 208, p = row - i * 208;
  const int tid = threadIdx.x;
  bf16* outA = kvA + (size_t)row * 768;
  bf16* outB = kvB + (size_t)row * 768;
  if (p >= 197) {
    for (int c = tid; c < 768; c += 256) { outA[c] = (bf16)0.0f; outB[c] = (bf16)0.0f; }
    return;
  }
  const float* x = tmp + ((size_t)i * 197 + p) * 768;
  __shared__ float red[256];
  float part = 0.0f;
  for (int c = tid; c < 768; c += 256) part += x[c];
  red[tid] = part; __syncthreads();
  for (int s = 128; s > 0; s >>= 1) { if (tid < s) red[tid] += red[tid + s]; __syncthreads(); }
  const float mean = red[0] / 768.0f;
  __syncthreads();
  part = 0.0f;
  for (int c = tid; c < 768; c += 256) { const float d = x[c] - mean; part += d * d; }
  red[tid] = part; __syncthreads();
  for (int s = 128; s > 0; s >>= 1) { if (tid < s) red[tid] += red[tid + s]; __syncthreads(); }
  const float rstd = rsqrtf(red[0] / 768.0f + LN_EPS);
  for (int c = tid; c < 768; c += 256) {
    const float y = (x[c] - mean) * rstd * lnw[c] + lnb[c];
    outA[c] = (bf16)(y + pos[(size_t)p * 768 + c]);
    outB[c] = (bf16)(y + pos[(size_t)(197 + p) * 768 + c]);
  }
}

// ---------------------------------------------------------------------------
// Per-head transposed V tiles: kvT[s][i][h][n(0..63)][t(0..207)] = kv{A,B}[i][t][h*64+n]
// so the value GEMM's B operand (V^T rows) is contiguous along K. grid = 384 blocks.
// ---------------------------------------------------------------------------
__global__ void transpose_kv(const bf16* __restrict__ kvA, const bf16* __restrict__ kvB,
                             bf16* __restrict__ kvT) {
  const int b = blockIdx.x;              // s*192 + i*12 + h
  const int s = b / 192;
  const int rem = b - s * 192;
  const int i = rem / 12, h = rem - i * 12;
  const bf16* src = (s == 0 ? kvA : kvB) + (size_t)i * 208 * 768 + h * 64;
  bf16* dst = kvT + ((((size_t)s * 16 + i) * 12 + h) * 64) * 208;
  for (int idx = threadIdx.x; idx < 64 * 208; idx += blockDim.x) {
    const int n = idx / 208, t = idx - n * 208;
    dst[idx] = src[(size_t)t * 768 + n];
  }
}

// ---------------------------------------------------------------------------
// Per-side flash attention half: for (side s, batch i, head h):
//   S = q_h @ kv^T            (16x208 via 13 N-tiles, K=64)
//   m,Z,E = softmax stats     (unnormalized, per row)
//   U = E @ V                 (16x64 via 4 N-tiles, K=224)
// One wave per block; grid = (16, 12, 2).
// ---------------------------------------------------------------------------
__global__ void attn_flash_half(const bf16* __restrict__ qpad, const bf16* __restrict__ kvA,
                                const bf16* __restrict__ kvB, const bf16* __restrict__ kvT,
                                float* __restrict__ U, float* __restrict__ Mst,
                                float* __restrict__ Zst) {
  const int i = blockIdx.x, h = blockIdx.y, s = blockIdx.z;
  const int lane = threadIdx.x;
  __shared__ float Sbuf[16 * 208];
  __shared__ bf16  Ebuf[16 * 224];
  const bf16* kv = (s == 0 ? kvA : kvB) + (size_t)i * 208 * 768 + h * 64;
  const bf16* qh = qpad + h * 64;

  // ---- scores: S[16,208] = q(16x64) @ kv(208x64)^T, q fragments hoisted,
  //      kv fragments double-buffered across the 13 N-tiles
  const v16bf aq0 = load_frag_a(qh, 768, lane);
  const v16bf aq1 = load_frag_a(qh + 32, 768, lane);
  const bf16* pkv = lane_base_b(kv, 768, lane);
  v16bf fb0 = frag_b_at(pkv);
  v16bf fb1 = frag_b_at(pkv + 32);
  for (int t = 0; t < 12; ++t) {
    v16bf nb0 = frag_b_at(pkv + (size_t)(t + 1) * 16 * 768);
    v16bf nb1 = frag_b_at(pkv + (size_t)(t + 1) * 16 * 768 + 32);
    v8f acc = {};
    acc = wmma_bf16(aq0, fb0, acc);
    acc = wmma_bf16(aq1, fb1, acc);
    const int col = t * 16 + (lane & 15);
    const int r0 = 8 * (lane >> 4);
#pragma unroll
    for (int r = 0; r < 8; ++r) Sbuf[(r0 + r) * 208 + col] = acc[r];
    fb0 = nb0; fb1 = nb1;
  }
  {  // last tile
    v8f acc = {};
    acc = wmma_bf16(aq0, fb0, acc);
    acc = wmma_bf16(aq1, fb1, acc);
    const int col = 12 * 16 + (lane & 15);
    const int r0 = 8 * (lane >> 4);
#pragma unroll
    for (int r = 0; r < 8; ++r) Sbuf[(r0 + r) * 208 + col] = acc[r];
  }
  __syncthreads();

  // ---- softmax stats (rows 0..7 real; rows 8..15 and cols 197..223 zeroed)
  const float scale = 0.125f;  // 1/sqrt(64)
  if (lane < 8) {
    float m = -3.0e38f;
    for (int c = 0; c < 197; ++c) m = fmaxf(m, Sbuf[lane * 208 + c] * scale);
    float z = 0.0f;
    for (int c = 0; c < 197; ++c) {
      const float e = __expf(Sbuf[lane * 208 + c] * scale - m);
      z += e;
      Ebuf[lane * 224 + c] = (bf16)e;
    }
    for (int c = 197; c < 224; ++c) Ebuf[lane * 224 + c] = (bf16)0.0f;
    const int sidx = ((s * 16 + i) * 12 + h) * 8 + lane;
    Mst[sidx] = m; Zst[sidx] = z;
  } else if (lane < 16) {
    for (int c = 0; c < 224; ++c) Ebuf[lane * 224 + c] = (bf16)0.0f;
  }
  __syncthreads();

  // ---- values: U[16,64] = E(16x224) @ V(224x64), B operand = V^T rows
  const bf16* Vt = kvT + ((((size_t)s * 16 + i) * 12 + h) * 64) * 208;
  float* Urow = U + ((((size_t)s * 16 + i) * 12 + h) * 8) * 64;
  for (int nt = 0; nt < 4; ++nt) {
    v8f acc = {};
    for (int ks = 0; ks < 224; ks += 32) {
      v16bf a = load_frag_a(Ebuf + ks, 224, lane);
      v16bf b = load_frag_b(Vt + (size_t)nt * 16 * 208 + ks, 208, lane);
      acc = wmma_bf16(a, b, acc);
    }
    if ((lane >> 4) == 0) {
      const int n = nt * 16 + (lane & 15);
#pragma unroll
      for (int r = 0; r < 8; ++r) Urow[(size_t)r * 64 + n] = acc[r];
    }
  }
}

// ---------------------------------------------------------------------------
// Pair merge: values[(i,j),q,:] = (U1*e^{m1-m} - U2*e^{m2-m}) / (Z1*e^{m1-m}+Z2*e^{m2-m})
// grid = 2048 rows.
// ---------------------------------------------------------------------------
__global__ void merge_values(const float* __restrict__ U, const float* __restrict__ Mst,
                             const float* __restrict__ Zst, bf16* __restrict__ values) {
  const int row = blockIdx.x;            // pair*8 + q
  const int pair = row >> 3, q = row & 7;
  const int i = pair >> 4, j = pair & 15;
  for (int c = threadIdx.x; c < 768; c += blockDim.x) {
    const int h = c >> 6, d = c & 63;
    const int s1 = ((0 * 16 + i) * 12 + h) * 8 + q;
    const int s2 = ((1 * 16 + j) * 12 + h) * 8 + q;
    const float m1 = Mst[s1], m2 = Mst[s2];
    const float m = fmaxf(m1, m2);
    const float a = __expf(m1 - m), b = __expf(m2 - m);
    const float z = Zst[s1] * a + Zst[s2] * b;
    const float v = (U[(size_t)s1 * 64 + d] * a - U[(size_t)s2 * 64 + d] * b) / z;
    values[(size_t)row * 768 + c] = (bf16)v;
  }
}

// ---------------------------------------------------------------------------
extern "C" void kernel_launch(void* const* d_in, const int* in_sizes, int n_in,
                              void* d_out, int out_size, void* d_ws, size_t ws_size,
                              hipStream_t stream) {
  const float* q_x    = (const float*)d_in[0];
  const float* kv_x   = (const float*)d_in[1];
  const float* pos    = (const float*)d_in[2];
  const float* q_w    = (const float*)d_in[3];
  const float* q_b    = (const float*)d_in[4];
  const float* kv_w   = (const float*)d_in[5];
  const float* kv_b   = (const float*)d_in[6];
  const float* out_w  = (const float*)d_in[7];
  const float* out_b  = (const float*)d_in[8];
  const float* lnq_w  = (const float*)d_in[9];
  const float* lnq_b  = (const float*)d_in[10];
  const float* lnkv_w = (const float*)d_in[11];
  const float* lnkv_b = (const float*)d_in[12];
  float* out = (float*)d_out;

  // kv-proj M padded from 3152 (=16*197) to 3168 so 32-row wave strips divide it
  const int MKV = 3168;

  char* ws = (char*)d_ws;
  size_t off = 0;
  auto alloc = [&](size_t bytes) -> char* {
    char* p = ws + off;
    off += (bytes + 255) & ~(size_t)255;
    return p;
  };
  bf16*  kvx_bf  = (bf16*)alloc((size_t)MKV * 768 * 2);
  bf16*  kvw_bf  = (bf16*)alloc((size_t)768 * 768 * 2);
  bf16*  outw_bf = (bf16*)alloc((size_t)768 * 768 * 2);
  float* tmp_kv  = (float*)alloc((size_t)MKV * 768 * 4);
  bf16*  qpad    = (bf16*)alloc((size_t)16 * 768 * 2);
  bf16*  kvA     = (bf16*)alloc((size_t)16 * 208 * 768 * 2);
  bf16*  kvB     = (bf16*)alloc((size_t)16 * 208 * 768 * 2);
  bf16*  kvT     = (bf16*)alloc((size_t)2 * 16 * 12 * 64 * 208 * 2);
  float* U       = (float*)alloc((size_t)2 * 16 * 12 * 8 * 64 * 4);
  float* Mst     = (float*)alloc((size_t)2 * 16 * 12 * 8 * 4);
  float* Zst     = (float*)alloc((size_t)2 * 16 * 12 * 8 * 4);
  bf16*  values  = (bf16*)alloc((size_t)2048 * 768 * 2);
  if (off > ws_size) return;

  // stage 0: bf16 copies of the big operands (zero-padding the kv_x tail rows)
  cvt_f32_to_bf16<<<512, 256, 0, stream>>>(kv_x, kvx_bf, 3152 * 768, MKV * 768);
  cvt_f32_to_bf16<<<512, 256, 0, stream>>>(kv_w, kvw_bf, 768 * 768, 768 * 768);
  cvt_f32_to_bf16<<<512, 256, 0, stream>>>(out_w, outw_bf, 768 * 768, 768 * 768);

  // stage 1: kv projection (WMMA GEMM, bias fused)
  {
    const int waves = (MKV / 32) * (768 / 64);
    gemm_bt_wmma<<<(waves + 7) / 8, 256, 0, stream>>>(kvx_bf, kvw_bf, kv_b, tmp_kv,
                                                      MKV, 768, 768);
  }

  // stage 2: q projection + LN (tiny), kv LN + positional bias
  qproj_ln<<<16, 256, 0, stream>>>(q_x, q_w, q_b, lnq_w, lnq_b, qpad);
  ln_pos<<<16 * 208, 256, 0, stream>>>(tmp_kv, lnkv_w, lnkv_b, pos, kvA, kvB);
  transpose_kv<<<384, 256, 0, stream>>>(kvA, kvB, kvT);

  // stage 3: 32 per-side flash-attention halves (WMMA scores + WMMA values)
  attn_flash_half<<<dim3(16, 12, 2), 32, 0, stream>>>(qpad, kvA, kvB, kvT, U, Mst, Zst);

  // stage 4: merge the 256 pairs from the 2x16 halves (cheap scalar math)
  merge_values<<<2048, 256, 0, stream>>>(U, Mst, Zst, values);

  // stage 5: output projection (WMMA GEMM, bias fused) -> d_out
  {
    const int waves = (2048 / 32) * (768 / 64);
    gemm_bt_wmma<<<(waves + 7) / 8, 256, 0, stream>>>(values, outw_bf, out_b, out,
                                                      2048, 768, 768);
  }
  (void)in_sizes; (void)n_in; (void)out_size;
}
